// mgn_GRUODECell_53987738911235
// MI455X (gfx1250) — matrix operand order repaired and checked by reading, not compiled, not run on hardware
//
#include <hip/hip_runtime.h>

typedef _Float16 v16h __attribute__((ext_vector_type(16)));
typedef float    v8f  __attribute__((ext_vector_type(8)));

#define ISZ   64
#define ND    32
#define HDIM  2048   // ISZ * ND

__device__ __forceinline__ float fast_sigmoid(float x) {
    // v_exp_f32 + v_rcp_f32 (no IEEE div sequence)
    return __builtin_amdgcn_rcpf(1.0f + __expf(-x));
}
__device__ __forceinline__ float fast_tanh(float x) {
    // tanh(x) = 1 - 2/(1 + exp(2x))
    return 1.0f - 2.0f * __builtin_amdgcn_rcpf(1.0f + __expf(2.0f * x));
}

__global__ __launch_bounds__(256) void gruode_wmma_f16(
    const float* __restrict__ h_in,
    const float* __restrict__ Ur, const float* __restrict__ Uz, const float* __restrict__ Uh,
    const float* __restrict__ br, const float* __restrict__ bz, const float* __restrict__ bh,
    float* __restrict__ dh_out, float* __restrict__ ht_out, int Btiles)
{
    // per-wave 16x32 f32 scratch for the C/D-layout -> A-layout remap of r
    __shared__ float lds_r[8][16 * ND];

    const int g     = blockIdx.x;          // group 0..63
    const int wave  = threadIdx.x >> 5;
    const int lane  = threadIdx.x & 31;
    const int m     = lane & 15;           // A: row M ; B: col N ; C/D: col N
    const int hi    = lane >> 4;           // lane half
    const int j0    = hi ? 8 : 0;          // A-layout K base (K: j0..j0+7, j0+16..j0+23)
    const int khalf = hi ? 16 : 0;         // unused; B rows also use j0 (same K pattern as A)

    // ---- B matrices resident in VGPRs: 6 halves (r0,r1,z0,z1,h0,h1) ----
    v16h Br0, Br1, Bz0, Bz1, Bh0, Bh1;
    {
        const int ubase = g * ND * ND;
        #pragma unroll
        for (int e = 0; e < 16; ++e) {
            const int row = ubase + (j0 + e + (e & 8)) * ND;
            Br0[e] = (_Float16)Ur[row + m];
            Br1[e] = (_Float16)Ur[row + 16 + m];
            Bz0[e] = (_Float16)Uz[row + m];
            Bz1[e] = (_Float16)Uz[row + 16 + m];
            Bh0[e] = (_Float16)Uh[row + m];
            Bh1[e] = (_Float16)Uh[row + 16 + m];
        }
    }
    const float bias_r0 = br[g * ND + m], bias_r1 = br[g * ND + 16 + m];
    const float bias_z0 = bz[g * ND + m], bias_z1 = bz[g * ND + 16 + m];
    const float bias_h0 = bh[g * ND + m], bias_h1 = bh[g * ND + 16 + m];

    float* __restrict__ myr = lds_r[wave];
    const v8f czero = {0.f, 0.f, 0.f, 0.f, 0.f, 0.f, 0.f, 0.f};

    const int waveStride = gridDim.y * 8;
    for (int t = blockIdx.y * 8 + wave; t < Btiles; t += waveStride) {
        const int b0 = t * 16;

        // ---- load A: 16 h values per lane (rows m, K pattern j0..j0+7, j0+16..j0+23)
        const float* hp = h_in + (size_t)(b0 + m) * HDIM + g * ND + j0;
        float hv[16];
        {
            const float4 f0 = *(const float4*)(hp);
            const float4 f1 = *(const float4*)(hp + 4);
            const float4 f2 = *(const float4*)(hp + 16);
            const float4 f3 = *(const float4*)(hp + 20);
            hv[0]=f0.x; hv[1]=f0.y; hv[2]=f0.z;  hv[3]=f0.w;
            hv[4]=f1.x; hv[5]=f1.y; hv[6]=f1.z;  hv[7]=f1.w;
            hv[8]=f2.x; hv[9]=f2.y; hv[10]=f2.z; hv[11]=f2.w;
            hv[12]=f3.x; hv[13]=f3.y; hv[14]=f3.z; hv[15]=f3.w;
        }
        // prefetch next tile's rows into near cache (speculative; dropped if OOB)
        __builtin_prefetch(hp + (size_t)waveStride * 16 * HDIM, 0, 3);

        v16h A;
        #pragma unroll
        for (int e = 0; e < 16; ++e) A[e] = (_Float16)hv[e];

        // ---- r and z gates: full K=32 in one WMMA each (two N-halves)
        v8f accr0 = __builtin_amdgcn_wmma_f32_16x16x32_f16(false, A, false, Br0, (short)0, czero, false, false);
        v8f accr1 = __builtin_amdgcn_wmma_f32_16x16x32_f16(false, A, false, Br1, (short)0, czero, false, false);
        v8f accz0 = __builtin_amdgcn_wmma_f32_16x16x32_f16(false, A, false, Bz0, (short)0, czero, false, false);
        v8f accz1 = __builtin_amdgcn_wmma_f32_16x16x32_f16(false, A, false, Bz1, (short)0, czero, false, false);

        v8f z0, z1;
        #pragma unroll
        for (int v = 0; v < 8; ++v) {
            const float r0 = fast_sigmoid(accr0[v] + bias_r0);
            const float r1 = fast_sigmoid(accr1[v] + bias_r1);
            z0[v] = fast_sigmoid(accz0[v] + bias_z0);
            z1[v] = fast_sigmoid(accz1[v] + bias_z1);
            // stash r in C/D layout: row = v + 8*hi, cols m and m+16
            const int row = (v + hi * 8) * ND;
            myr[row + m]      = r0;
            myr[row + 16 + m] = r1;
        }

        // same-wave DS ops are in-order; fence the compiler + counters
        asm volatile("s_wait_dscnt 0" ::: "memory");

        // ---- gather r in A layout, fuse r*h, convert to f16
        v16h A2;
        {
            const float* rp = myr + m * ND + j0;
            const float4 r0v = *(const float4*)(rp);
            const float4 r1v = *(const float4*)(rp + 4);
            const float4 r2v = *(const float4*)(rp + 16);
            const float4 r3v = *(const float4*)(rp + 20);
            float rv[16];
            rv[0]=r0v.x; rv[1]=r0v.y; rv[2]=r0v.z;  rv[3]=r0v.w;
            rv[4]=r1v.x; rv[5]=r1v.y; rv[6]=r1v.z;  rv[7]=r1v.w;
            rv[8]=r2v.x; rv[9]=r2v.y; rv[10]=r2v.z; rv[11]=r2v.w;
            rv[12]=r3v.x; rv[13]=r3v.y; rv[14]=r3v.z; rv[15]=r3v.w;
            #pragma unroll
            for (int e = 0; e < 16; ++e) A2[e] = (_Float16)(rv[e] * hv[e]);
        }

        // ---- candidate gate
        v8f acch0 = __builtin_amdgcn_wmma_f32_16x16x32_f16(false, A2, false, Bh0, (short)0, czero, false, false);
        v8f acch1 = __builtin_amdgcn_wmma_f32_16x16x32_f16(false, A2, false, Bh1, (short)0, czero, false, false);

        // ---- epilogue: dh = (1-z)*(h_tilde - h); outputs in C/D layout
        const int rowbase = b0 + hi * 8;
        #pragma unroll
        for (int v = 0; v < 8; ++v) {
            const size_t idx0 = (size_t)(rowbase + v) * HDIM + g * ND + m;
            const float hg0 = h_in[idx0];
            const float ht0 = fast_tanh(acch0[v] + bias_h0);
            dh_out[idx0] = (1.0f - z0[v]) * (ht0 - hg0);
            ht_out[idx0] = ht0;

            const size_t idx1 = idx0 + 16;
            const float hg1 = h_in[idx1];
            const float ht1 = fast_tanh(acch1[v] + bias_h1);
            dh_out[idx1] = (1.0f - z1[v]) * (ht1 - hg1);
            ht_out[idx1] = ht1;
        }
    }
}

extern "C" void kernel_launch(void* const* d_in, const int* in_sizes, int n_in,
                              void* d_out, int out_size, void* d_ws, size_t ws_size,
                              hipStream_t stream) {
    // inputs: 0=t (unused), 1=mgn_h, 2=U_r, 3=U_z, 4=U_h, 5=b_r, 6=b_z, 7=b_h
    const float* h  = (const float*)d_in[1];
    const float* Ur = (const float*)d_in[2];
    const float* Uz = (const float*)d_in[3];
    const float* Uh = (const float*)d_in[4];
    const float* br = (const float*)d_in[5];
    const float* bz = (const float*)d_in[6];
    const float* bh = (const float*)d_in[7];

    const int Btot  = in_sizes[1] / HDIM;   // 16384
    const int tiles = Btot / 16;            // 1024

    float* dh = (float*)d_out;
    float* ht = (float*)d_out + (size_t)Btot * HDIM;

    dim3 grid(ISZ, 32);   // 64 groups x 32 blocks -> 256 waves/group, 4 tiles each
    dim3 block(256);      // 8 waves
    gruode_wmma_f16<<<grid, block, 0, stream>>>(h, Ur, Uz, Uh, br, bz, bh, dh, ht, tiles);
}
